// GLAAttention_13778255085615
// MI455X (gfx1250) — matrix operand order repaired and checked by reading, not compiled, not run on hardware
//
#include <hip/hip_runtime.h>
#include <hip/hip_bf16.h>
#include <cstddef>

// ---------------- problem constants (match reference) ----------------
#define DIM   1024
#define HEADS 16
#define KDIM  512          // DIM/2
#define VDIM  1024
#define DK    32           // KDIM/HEADS
#define DV    64           // VDIM/HEADS
#define LOW   16
#define NORMALIZER 16.0f
#define EPS   1e-5f
#define BATCH 2
#define TLEN  4096
#define BT    (BATCH * TLEN)   // 8192 rows

typedef __attribute__((ext_vector_type(16))) _Float16 v16h;
typedef __attribute__((ext_vector_type(8)))  float    v8f;

// ---------------------------------------------------------------------
// f32 -> f16 elementwise convert (activations, row-major preserved)
// ---------------------------------------------------------------------
__global__ void cvt_f32_to_f16(const float* __restrict__ src,
                               _Float16* __restrict__ dst, int n) {
    int i = blockIdx.x * blockDim.x + threadIdx.x;
    if (i < n) dst[i] = (_Float16)src[i];
}

// ---------------------------------------------------------------------
// f32 (K x N, row-major) -> f16 transposed (N x K, row-major).
// Makes WMMA B-fragment loads fully contiguous (2 x b128 per lane).
// ---------------------------------------------------------------------
__global__ void cvt_f32_to_f16_T(const float* __restrict__ src,
                                 _Float16* __restrict__ dst, int K, int N) {
    size_t i = (size_t)blockIdx.x * blockDim.x + threadIdx.x;
    if (i >= (size_t)K * N) return;
    int k = (int)(i / N);
    int n = (int)(i % N);
    dst[(size_t)n * K + k] = (_Float16)src[i];
}

// ---------------------------------------------------------------------
// WMMA GEMM: C(MxN, f32, row-major) = A(MxK, f16, row-major) @ B,
// B supplied TRANSPOSED: Bt(N x K, f16, row-major).
// One wave computes a (16*MT) x (16*NT) C tile -> MT*NT WMMAs per
// 32-deep K step, software-pipelined with two explicit fragment sets
// (loads for the next K-step issue before the current WMMA group).
// All fragment loads are contiguous b128. Requires M%(16*MT)==0,
// N%(16*NT)==0, K%64==0 (holds at every call site), so EXEC is
// all-ones for each WMMA as the ISA requires.
// ---------------------------------------------------------------------
template <int MT, int NT>
__global__ __launch_bounds__(32)
void wmma_gemm_f16(const _Float16* __restrict__ A,
                   const _Float16* __restrict__ Bt,
                   float* __restrict__ C,
                   int M, int N, int K) {
    const int tileM = blockIdx.x;            // M/(16*MT) tiles
    const int tileN = blockIdx.y;            // N/(16*NT) tiles
    const int lane  = threadIdx.x;           // 0..31 (wave32)
    const int rowA  = tileM * 16 * MT;
    const int colB  = tileN * 16 * NT;

    // A frag: lane holds row m = lane&15; K = kb + 0..15 contiguous
    // with kb = 16*(lane>=16) -> two contiguous 8-half runs.
    const int m    = lane & 15;
    const int kb   = (lane >> 4) * 16;
    // B frag: lane holds col n = lane&15; K = koff + 0..15 contiguous in Bt.
    const int n    = lane & 15;
    const int koff = (lane >> 4) * 16;

    union Frag { v16h v; uint4 q[2]; };

    const _Float16* arow[MT];
    #pragma unroll
    for (int t = 0; t < MT; ++t)
        arow[t] = A + (size_t)(rowA + t * 16 + m) * K;
    const _Float16* brow[NT];
    #pragma unroll
    for (int t = 0; t < NT; ++t)
        brow[t] = Bt + (size_t)(colB + t * 16 + n) * K;

    v8f acc[MT][NT];
    #pragma unroll
    for (int i = 0; i < MT; ++i)
        #pragma unroll
        for (int j = 0; j < NT; ++j) acc[i][j] = (v8f){};

    Frag fa[2][MT], fb[2][NT];

    auto loadFrags = [&](int set, int k0) {
        #pragma unroll
        for (int i = 0; i < MT; ++i) {
            fa[set][i].q[0] = *(const uint4*)(arow[i] + k0 + kb);
            fa[set][i].q[1] = *(const uint4*)(arow[i] + k0 + kb + 8);
        }
        #pragma unroll
        for (int j = 0; j < NT; ++j) {
            fb[set][j].q[0] = *(const uint4*)(brow[j] + k0 + koff);
            fb[set][j].q[1] = *(const uint4*)(brow[j] + k0 + koff + 8);
        }
    };
    auto mmaSet = [&](int set) {
        #pragma unroll
        for (int i = 0; i < MT; ++i)
            #pragma unroll
            for (int j = 0; j < NT; ++j)
                acc[i][j] = __builtin_amdgcn_wmma_f32_16x16x32_f16(
                    false, fa[set][i].v, false, fb[set][j].v,
                    (short)0, acc[i][j], false, false);
    };

    // software pipeline: two 32-deep K steps per iteration, K%64==0
    loadFrags(0, 0);
    for (int k0 = 0; k0 < K; k0 += 64) {
        loadFrags(1, k0 + 32);                 // next set in flight
        mmaSet(0);
        if (k0 + 64 < K) loadFrags(0, k0 + 64);
        mmaSet(1);
    }

    // C/D layout: VGPR r -> row r (lanes 0-15) / row 8+r (lanes 16-31)
    const int m0 = (lane >= 16) ? 8 : 0;
    #pragma unroll
    for (int i = 0; i < MT; ++i) {
        #pragma unroll
        for (int j = 0; j < NT; ++j) {
            float* cbase = C + (size_t)(rowA + i * 16 + m0) * N
                             + colB + j * 16 + (lane & 15);
            #pragma unroll
            for (int r = 0; r < 8; ++r) cbase[(size_t)r * N] = acc[i][j][r];
        }
    }
}

// ---------------------------------------------------------------------
// low-rank gate stage 2: gk = logsigmoid(tmp @ Wgk2 + b) / NORMALIZER
// tmp: (BT, LOW) f32 ; Wgk2: (LOW, KDIM) ; gk: (BT, KDIM). K=16 -> VALU.
// ---------------------------------------------------------------------
__global__ void gk_lowrank(const float* __restrict__ tmp,
                           const float* __restrict__ W2,
                           const float* __restrict__ bias,
                           float* __restrict__ gk) {
    size_t i = (size_t)blockIdx.x * blockDim.x + threadIdx.x;
    if (i >= (size_t)BT * KDIM) return;
    int row = (int)(i / KDIM);
    int col = (int)(i % KDIM);
    float s = bias[col];
    #pragma unroll
    for (int r = 0; r < LOW; ++r)
        s += tmp[(size_t)row * LOW + r] * W2[(size_t)r * KDIM + col];
    // logsigmoid(x) = min(x,0) - log1p(exp(-|x|))
    float ls = fminf(s, 0.f) - __logf(1.f + __expf(-fabsf(s)));
    gk[i] = ls * (1.0f / NORMALIZER);
}

// ---------------------------------------------------------------------
// GLA recurrent scan. One block (64 threads = 2 wave32) per (b,h).
// Each thread owns one v-column j = tid of the 32x64 state, in VGPRs.
// ---------------------------------------------------------------------
__global__ __launch_bounds__(64)
void gla_scan(const float* __restrict__ q,
              const float* __restrict__ k,
              const float* __restrict__ gk,
              const float* __restrict__ v,
              float* __restrict__ o) {
    const int bh = blockIdx.x;
    const int b  = bh / HEADS;
    const int h  = bh % HEADS;
    const int tid = threadIdx.x;     // 0..63

    __shared__ float sq[DK], sk[DK], se[DK], sv[DV];

    float st[DK];
    #pragma unroll
    for (int i = 0; i < DK; ++i) st[i] = 0.f;

    const float scale = 0.17677669529663687f;  // DK^-0.5

    for (int t = 0; t < TLEN; ++t) {
        const size_t baseK = ((size_t)b * TLEN + t) * KDIM + (size_t)h * DK;
        const size_t baseV = ((size_t)b * TLEN + t) * VDIM + (size_t)h * DV;
        if (tid < DK) {
            sq[tid] = q[baseK + tid];
            sk[tid] = k[baseK + tid];
            se[tid] = __expf(gk[baseK + tid]);
        }
        sv[tid] = v[baseV + tid];
        __syncthreads();

        const float vj = sv[tid];
        float acc = 0.f;
        #pragma unroll
        for (int kk = 0; kk < DK; ++kk) {
            st[kk] = st[kk] * se[kk] + sk[kk] * vj;
            acc += sq[kk] * st[kk];
        }
        o[baseV + tid] = acc * scale;
        __syncthreads();
    }
}

// ---------------------------------------------------------------------
// fused per-head RMSNorm + swish output gate; writes f16 for final GEMM.
// One wave32 per head (DV=64 -> 2 elems/lane), shfl_xor reduction.
// ---------------------------------------------------------------------
__global__ __launch_bounds__(256)
void gate_norm(const float* __restrict__ o,
               const float* __restrict__ g,
               const float* __restrict__ gnw,
               _Float16* __restrict__ out) {
    const int lane = threadIdx.x & 31;
    const int warp = threadIdx.x >> 5;
    const size_t head = (size_t)blockIdx.x * 8 + warp;   // B*T*H heads total
    const size_t base = head * DV;

    float o0 = o[base + lane];
    float o1 = o[base + lane + 32];
    float ss = o0 * o0 + o1 * o1;
    #pragma unroll
    for (int mm = 16; mm > 0; mm >>= 1) ss += __shfl_xor(ss, mm, 32);
    const float r = rsqrtf(ss * (1.0f / DV) + EPS);

    float g0 = g[base + lane];
    float g1 = g[base + lane + 32];
    g0 = g0 * (1.f / (1.f + __expf(-g0)));   // swish
    g1 = g1 * (1.f / (1.f + __expf(-g1)));

    out[base + lane]      = (_Float16)(o0 * r * gnw[lane]      * g0);
    out[base + lane + 32] = (_Float16)(o1 * r * gnw[lane + 32] * g1);
}

// ---------------------------------------------------------------------
// launch
// inputs: 0:x 1:Wq 2:Wk 3:Wv 4:Wgk1 5:Wgk2 6:bgk2 7:Wg 8:gnw 9:Wo
// ---------------------------------------------------------------------
extern "C" void kernel_launch(void* const* d_in, const int* in_sizes, int n_in,
                              void* d_out, int out_size, void* d_ws, size_t ws_size,
                              hipStream_t stream) {
    const float* x    = (const float*)d_in[0];
    const float* Wq   = (const float*)d_in[1];
    const float* Wk   = (const float*)d_in[2];
    const float* Wv   = (const float*)d_in[3];
    const float* Wgk1 = (const float*)d_in[4];
    const float* Wgk2 = (const float*)d_in[5];
    const float* bgk2 = (const float*)d_in[6];
    const float* Wg   = (const float*)d_in[7];
    const float* gnw  = (const float*)d_in[8];
    const float* Wo   = (const float*)d_in[9];
    float* outp = (float*)d_out;

    // ---- workspace layout (256B aligned) ----
    char* ws = (char*)d_ws;
    size_t off = 0;
    auto alloc = [&](size_t bytes) -> char* {
        char* p = ws + off;
        off = (off + bytes + 255) & ~(size_t)255;
        return p;
    };
    _Float16* xh    = (_Float16*)alloc((size_t)BT * DIM * 2);
    _Float16* wqh   = (_Float16*)alloc((size_t)DIM * KDIM * 2);   // transposed
    _Float16* wkh   = (_Float16*)alloc((size_t)DIM * KDIM * 2);   // transposed
    _Float16* wvh   = (_Float16*)alloc((size_t)DIM * VDIM * 2);   // transposed
    _Float16* wgh   = (_Float16*)alloc((size_t)DIM * VDIM * 2);   // transposed
    _Float16* wgk1h = (_Float16*)alloc((size_t)DIM * LOW * 2);    // transposed
    _Float16* woh   = (_Float16*)alloc((size_t)VDIM * DIM * 2);   // transposed
    float* qb   = (float*)alloc((size_t)BT * KDIM * 4);
    float* kbuf = (float*)alloc((size_t)BT * KDIM * 4);
    float* gkb  = (float*)alloc((size_t)BT * KDIM * 4);
    float* vb   = (float*)alloc((size_t)BT * VDIM * 4);
    float* gb   = (float*)alloc((size_t)BT * VDIM * 4);
    float* tmpb = (float*)alloc((size_t)BT * LOW * 4);
    float* ob   = (float*)alloc((size_t)BT * VDIM * 4);
    // gated f16 output reuses the (now dead) q buffer: 16MB <= 16MB
    _Float16* ogh = (_Float16*)qb;
    (void)ws_size; (void)n_in; (void)in_sizes; (void)out_size;

    // ---- 1) conversions: x row-major f16; weights transposed f16 ----
    cvt_f32_to_f16<<<(BT * DIM + 255) / 256, 256, 0, stream>>>(x, xh, BT * DIM);
    auto cvtT = [&](const float* s, _Float16* d, int K, int N) {
        int n = K * N;
        cvt_f32_to_f16_T<<<(n + 255) / 256, 256, 0, stream>>>(s, d, K, N);
    };
    cvtT(Wq,   wqh,   DIM, KDIM);
    cvtT(Wk,   wkh,   DIM, KDIM);
    cvtT(Wv,   wvh,   DIM, VDIM);
    cvtT(Wg,   wgh,   DIM, VDIM);
    cvtT(Wgk1, wgk1h, DIM, LOW);
    cvtT(Wo,   woh,   VDIM, DIM);

    // ---- 2) WMMA projections (32x64 tile per wave, MT=2 x NT=4) ----
    auto gemm = [&](const _Float16* A, const _Float16* B, float* C,
                    int M, int N, int K) {
        dim3 grid(M / 32, N / 64);
        wmma_gemm_f16<2, 4><<<grid, 32, 0, stream>>>(A, B, C, M, N, K);
    };
    gemm(xh, wqh, qb,   BT, KDIM, DIM);   // q
    gemm(xh, wkh, kbuf, BT, KDIM, DIM);   // k
    gemm(xh, wvh, vb,   BT, VDIM, DIM);   // v
    gemm(xh, wgh, gb,   BT, VDIM, DIM);   // g
    {   // low-rank stage 1: N=16 -> single-tile variant
        dim3 grid(BT / 16, 1);
        wmma_gemm_f16<1, 1><<<grid, 32, 0, stream>>>(xh, wgk1h, tmpb, BT, LOW, DIM);
    }

    // ---- 3) low-rank gate stage 2 + logsigmoid ----
    {
        size_t n = (size_t)BT * KDIM;
        gk_lowrank<<<(unsigned)((n + 255) / 256), 256, 0, stream>>>(tmpb, Wgk2, bgk2, gkb);
    }

    // ---- 4) recurrent GLA scan: one block per (b,h) ----
    gla_scan<<<BATCH * HEADS, 64, 0, stream>>>(qb, kbuf, gkb, vb, ob);

    // ---- 5) fused RMSNorm + swish gate -> f16 ----
    gate_norm<<<(BATCH * TLEN * HEADS) / 8, 256, 0, stream>>>(ob, gb, gnw, ogh);

    // ---- 6) final projection: out = gated @ Wo ----
    gemm(ogh, woh, outp, BT, DIM, VDIM);
}
